// ProLMSelfAttention_53772990546412
// MI455X (gfx1250) — matrix operand order repaired and checked by reading, not compiled, not run on hardware
//
#include <hip/hip_runtime.h>
#include <hip/hip_bf16.h>

// Problem constants (from reference)
#define BB 2
#define SS 2048
#define DMODEL 1024
#define NH 16
#define DH 64

typedef float v8f  __attribute__((ext_vector_type(8)));
typedef __bf16 v16bf __attribute__((ext_vector_type(16)));

union Frag { uint4 u[2]; v16bf f; };

__device__ inline void lds_fence() { asm volatile("" ::: "memory"); }

__device__ inline v8f zero8() {
  v8f z = {0.f,0.f,0.f,0.f,0.f,0.f,0.f,0.f};
  return z;
}

__device__ inline v16bf cvt16(float4 a, float4 b, float4 c, float4 d) {
  v16bf f;
  f[0]=(__bf16)a.x;  f[1]=(__bf16)a.y;  f[2]=(__bf16)a.z;  f[3]=(__bf16)a.w;
  f[4]=(__bf16)b.x;  f[5]=(__bf16)b.y;  f[6]=(__bf16)b.z;  f[7]=(__bf16)b.w;
  f[8]=(__bf16)c.x;  f[9]=(__bf16)c.y;  f[10]=(__bf16)c.z; f[11]=(__bf16)c.w;
  f[12]=(__bf16)d.x; f[13]=(__bf16)d.y; f[14]=(__bf16)d.z; f[15]=(__bf16)d.w;
  return f;
}

// ---------------------------------------------------------------------------
// Kernel 1: QKV projection (bf16 WMMA, f32 accum) + scale + RoPE, storing
// Q as A-fragments, K as B-fragments of K^T, V as B-fragments, all bf16.
//   q_ws: [bh][S/16 tile][2 dstep]  x 512 bf16 fragments
//   k_ws: [bh][S/16 tile][2 dstep]  x 512 bf16 fragments
//   v_ws: [bh][S/32 kstep][4 nsub]  x 512 bf16 fragments
// ---------------------------------------------------------------------------
__global__ __launch_bounds__(128) void qkv_rope_kernel(
    const float* __restrict__ hidden, const int* __restrict__ pos_ids,
    const float* __restrict__ Wq, const float* __restrict__ Wk,
    const float* __restrict__ Wv,
    uint4* __restrict__ q_ws, uint4* __restrict__ k_ws, uint4* __restrict__ v_ws)
{
  __shared__ alignas(32) __bf16 lds[4][1024];   // 16x64 bf16 tile per wave

  const int tid  = threadIdx.x;
  const int w    = tid >> 5;
  const int lane = tid & 31;
  const int half = lane >> 4;
  const int ln   = lane & 15;

  const int mt32 = blockIdx.x * 4 + w;     // 0..127, 32 rows each
  const int h    = blockIdx.y;             // head
  const int n0   = h * DH;
  const int bb   = (mt32 * 32) / SS;       // batch
  const int sbase = (mt32 * 32) % SS;      // seq row base within batch
  const int bh   = bb * NH + h;

  const float* Ws[3] = {Wq, Wk, Wv};

  v8f acc[3][2][4];
  #pragma unroll
  for (int m = 0; m < 3; ++m)
    #pragma unroll
    for (int s = 0; s < 2; ++s)
      #pragma unroll
      for (int n = 0; n < 4; ++n)
        acc[m][s][n] = zero8();

  for (int k0 = 0; k0 < DMODEL; k0 += 32) {
    // A-fragments of hidden for both 16-row subtiles
    v16bf af[2];
    #pragma unroll
    for (int ms = 0; ms < 2; ++ms) {
      const float* src = hidden + (size_t)(mt32 * 32 + ms * 16 + ln) * DMODEL
                         + k0 + 8 * half;
      float4 s0 = *(const float4*)(src);
      float4 s1 = *(const float4*)(src + 4);
      float4 s2 = *(const float4*)(src + 16);
      float4 s3 = *(const float4*)(src + 20);
      af[ms] = cvt16(s0, s1, s2, s3);
    }
    // B-fragments of the three weight matrices (used transposed: B[k][m]=W[m][k])
    #pragma unroll
    for (int mat = 0; mat < 3; ++mat) {
      #pragma unroll
      for (int ns = 0; ns < 4; ++ns) {
        const float* wsrc = Ws[mat] + (size_t)(n0 + ns * 16 + ln) * DMODEL
                            + k0 + 8 * half;
        float4 w0 = *(const float4*)(wsrc);
        float4 w1 = *(const float4*)(wsrc + 4);
        float4 w2 = *(const float4*)(wsrc + 16);
        float4 w3 = *(const float4*)(wsrc + 20);
        v16bf bf = cvt16(w0, w1, w2, w3);
        #pragma unroll
        for (int ms = 0; ms < 2; ++ms)
          acc[mat][ms][ns] = __builtin_amdgcn_wmma_f32_16x16x32_bf16(
              false, af[ms], false, bf, (short)0, acc[mat][ms][ns], false, false);
      }
    }
  }

  // RoPE frequencies: inv_freq[f] = 10000^(-f/32), f = a*16 + ln
  const float L = 13.287712379549449f / 32.0f;   // log2(10000)/32
  const float inv0 = exp2f(-(float)(ln)      * L);
  const float inv1 = exp2f(-(float)(ln + 16) * L);

  const uint4* lp = (const uint4*)&lds[w][0];

  #pragma unroll
  for (int ms = 0; ms < 2; ++ms) {
    const int t16 = sbase / 16 + ms;               // q/kv tile index in [0,128)
    int pr[8];
    #pragma unroll
    for (int r = 0; r < 8; ++r)
      pr[r] = pos_ids[bb * SS + sbase + ms * 16 + r + 8 * half];

    // ---- Q pass: scale + RoPE -> LDS [s][d] -> A-fragment store ----
    #pragma unroll
    for (int a = 0; a < 2; ++a) {
      const float inv = (a == 0) ? inv0 : inv1;
      #pragma unroll
      for (int r = 0; r < 8; ++r) {
        float x1 = acc[0][ms][a][r]     * 0.125f;
        float x2 = acc[0][ms][a + 2][r] * 0.125f;
        float ang = (float)pr[r] * inv;
        float c = __cosf(ang), s = __sinf(ang);
        lds[w][(r + 8 * half) * 64 + a * 16 + ln]       = (__bf16)(x1 * c - x2 * s);
        lds[w][(r + 8 * half) * 64 + (a + 2) * 16 + ln] = (__bf16)(x2 * c + x1 * s);
      }
    }
    lds_fence();
    #pragma unroll
    for (int ds = 0; ds < 2; ++ds) {
      uint4 u0 = lp[ln * 8 + ds * 4 + half];
      uint4 u1 = lp[ln * 8 + ds * 4 + half + 2];
      uint4* dst = q_ws + ((size_t)(bh * 128 + t16) * 2 + ds) * 64 + lane * 2;
      dst[0] = u0; dst[1] = u1;
    }
    lds_fence();

    // ---- K pass: RoPE -> LDS [s][d] -> B-fragment (of K^T) store ----
    #pragma unroll
    for (int a = 0; a < 2; ++a) {
      const float inv = (a == 0) ? inv0 : inv1;
      #pragma unroll
      for (int r = 0; r < 8; ++r) {
        float x1 = acc[1][ms][a][r];
        float x2 = acc[1][ms][a + 2][r];
        float ang = (float)pr[r] * inv;
        float c = __cosf(ang), s = __sinf(ang);
        lds[w][(r + 8 * half) * 64 + a * 16 + ln]       = (__bf16)(x1 * c - x2 * s);
        lds[w][(r + 8 * half) * 64 + (a + 2) * 16 + ln] = (__bf16)(x2 * c + x1 * s);
      }
    }
    lds_fence();
    #pragma unroll
    for (int ds = 0; ds < 2; ++ds) {
      uint4 u0 = lp[ln * 8 + ds * 4 + half];
      uint4 u1 = lp[ln * 8 + ds * 4 + half + 2];
      uint4* dst = k_ws + ((size_t)(bh * 128 + t16) * 2 + ds) * 64 + lane * 2;
      dst[0] = u0; dst[1] = u1;
    }
    lds_fence();

    // ---- V pass: -> LDS transposed [d][s] -> B-fragment store ----
    #pragma unroll
    for (int a = 0; a < 4; ++a)
      #pragma unroll
      for (int r = 0; r < 8; ++r)
        lds[w][(a * 16 + ln) * 16 + r + 8 * half] = (__bf16)acc[2][ms][a][r];
    lds_fence();
    {
      const int half_t = t16 & 1;           // which 16-row half of the 32-kv fragment
      if (half == half_t) {
        #pragma unroll
        for (int ns = 0; ns < 4; ++ns) {
          uint4 u0 = lp[(ns * 16 + ln) * 2];
          uint4 u1 = lp[(ns * 16 + ln) * 2 + 1];
          uint4* dst = v_ws + ((size_t)(bh * 64 + (t16 >> 1)) * 4 + ns) * 64
                       + (half_t * 16 + ln) * 2;
          dst[0] = u0; dst[1] = u1;
        }
      }
    }
    lds_fence();
  }
}

// ---------------------------------------------------------------------------
// Kernel 2: flash attention. One wave = 16 q rows x one (b,h), full KV sweep.
// ---------------------------------------------------------------------------
__global__ __launch_bounds__(128) void attn_kernel(
    const uint4* __restrict__ q_ws, const uint4* __restrict__ k_ws,
    const uint4* __restrict__ v_ws, const float* __restrict__ mask,
    float* __restrict__ out)
{
  __shared__ alignas(32) __bf16 plds[4][512];    // 16x32 bf16 P tile per wave

  const int tid  = threadIdx.x;
  const int w    = tid >> 5;
  const int lane = tid & 31;
  const int half = lane >> 4;
  const int ln   = lane & 15;

  const int qt = blockIdx.x * 4 + w;       // 0..127
  const int bh = blockIdx.y;
  const int b  = bh >> 4;
  const int h  = bh & 15;
  const int q0 = qt * 16;

  // Q A-fragments (resident for the whole sweep)
  Frag qf[2];
  #pragma unroll
  for (int ds = 0; ds < 2; ++ds) {
    const uint4* src = q_ws + ((size_t)(bh * 128 + qt) * 2 + ds) * 64 + lane * 2;
    qf[ds].u[0] = src[0]; qf[ds].u[1] = src[1];
  }

  v8f ctx[4];
  #pragma unroll
  for (int n = 0; n < 4; ++n) ctx[n] = zero8();
  float mrun[8], lrun[8];
  #pragma unroll
  for (int r = 0; r < 8; ++r) { mrun[r] = -3.0e38f; lrun[r] = 0.f; }

  const uint4* pp = (const uint4*)&plds[w][0];

  for (int kv0 = 0; kv0 < SS; kv0 += 32) {
    // ---- scores: two 16x16 tiles via QK^T ----
    v8f sc[2];
    #pragma unroll
    for (int sub = 0; sub < 2; ++sub) {
      v8f a = zero8();
      #pragma unroll
      for (int ds = 0; ds < 2; ++ds) {
        Frag kf;
        const uint4* ks = k_ws + ((size_t)(bh * 128 + (kv0 >> 4) + sub) * 2 + ds) * 64
                          + lane * 2;
        kf.u[0] = ks[0]; kf.u[1] = ks[1];
        a = __builtin_amdgcn_wmma_f32_16x16x32_bf16(
                false, qf[ds].f, false, kf.f, (short)0, a, false, false);
      }
      const float* mrow = mask + (size_t)b * SS * SS
                          + (size_t)(q0 + 8 * half) * SS + kv0 + sub * 16 + ln;
      #pragma unroll
      for (int r = 0; r < 8; ++r) a[r] += mrow[(size_t)r * SS];
      sc[sub] = a;
    }

    // ---- online softmax (row stats across the 16-lane group) ----
    float alpha[8];
    #pragma unroll
    for (int r = 0; r < 8; ++r) {
      float mx = fmaxf(sc[0][r], sc[1][r]);
      #pragma unroll
      for (int off = 1; off < 16; off <<= 1)
        mx = fmaxf(mx, __shfl_xor(mx, off, 32));
      float mnew = fmaxf(mrun[r], mx);
      alpha[r] = __expf(mrun[r] - mnew);
      mrun[r] = mnew;
      float p0 = __expf(sc[0][r] - mnew);
      float p1 = __expf(sc[1][r] - mnew);
      sc[0][r] = p0; sc[1][r] = p1;
      float rs = p0 + p1;
      #pragma unroll
      for (int off = 1; off < 16; off <<= 1)
        rs += __shfl_xor(rs, off, 32);
      lrun[r] = lrun[r] * alpha[r] + rs;
    }
    #pragma unroll
    for (int ns = 0; ns < 4; ++ns)
      #pragma unroll
      for (int r = 0; r < 8; ++r) ctx[ns][r] *= alpha[r];

    // ---- P (16x32) -> LDS -> A-fragment ----
    #pragma unroll
    for (int sub = 0; sub < 2; ++sub)
      #pragma unroll
      for (int r = 0; r < 8; ++r)
        plds[w][(r + 8 * half) * 32 + sub * 16 + ln] = (__bf16)sc[sub][r];
    lds_fence();
    Frag pf;
    pf.u[0] = pp[ln * 4 + half * 2];
    pf.u[1] = pp[ln * 4 + half * 2 + 1];
    lds_fence();

    // ---- ctx += P x V ----
    #pragma unroll
    for (int ns = 0; ns < 4; ++ns) {
      Frag vf;
      const uint4* vs = v_ws + ((size_t)(bh * 64 + (kv0 >> 5)) * 4 + ns) * 64 + lane * 2;
      vf.u[0] = vs[0]; vf.u[1] = vs[1];
      ctx[ns] = __builtin_amdgcn_wmma_f32_16x16x32_bf16(
                    false, pf.f, false, vf.f, (short)0, ctx[ns], false, false);
    }
  }

  // ---- normalize + store fp32 output [B,S,DM] ----
  #pragma unroll
  for (int ns = 0; ns < 4; ++ns)
    #pragma unroll
    for (int r = 0; r < 8; ++r) {
      float v = ctx[ns][r] / lrun[r];
      out[(size_t)(b * SS + q0 + r + 8 * half) * DMODEL + h * DH + ns * 16 + ln] = v;
    }
}

extern "C" void kernel_launch(void* const* d_in, const int* in_sizes, int n_in,
                              void* d_out, int out_size, void* d_ws, size_t ws_size,
                              hipStream_t stream) {
  (void)in_sizes; (void)n_in; (void)out_size; (void)ws_size;
  const float* hidden = (const float*)d_in[0];
  const float* mask   = (const float*)d_in[1];
  const int*   pos    = (const int*)d_in[2];
  const float* Wq     = (const float*)d_in[3];
  const float* Wk     = (const float*)d_in[4];
  const float* Wv     = (const float*)d_in[5];

  // workspace: 8 MB each for Q/K/V bf16 fragment stores (24 MB total)
  uint4* q_ws = (uint4*)d_ws;
  uint4* k_ws = (uint4*)((char*)d_ws + (8u << 20));
  uint4* v_ws = (uint4*)((char*)d_ws + (16u << 20));

  qkv_rope_kernel<<<dim3(32, 16), 128, 0, stream>>>(
      hidden, pos, Wq, Wk, Wv, q_ws, k_ws, v_ws);
  attn_kernel<<<dim3(32, 32), 128, 0, stream>>>(
      q_ws, k_ws, v_ws, mask, (float*)d_out);
}
